// LatentBatchedExperts_12773232738934
// MI455X (gfx1250) — matrix operand order repaired and checked by reading, not compiled, not run on hardware
//
#include <hip/hip_runtime.h>
#include <hip/hip_bf16.h>

// ---------------------------------------------------------------------------
// Problem constants (match reference: T=8192, E=8, L=256, F=1024)
// ---------------------------------------------------------------------------
#define T_TOK 8192
#define N_EXP 8
#define L_DIM 256
#define F_DIM 1024
#define TILE_M 16
#define M_TILES 2
#define TOK_WG (TILE_M * M_TILES)            // 32 tokens per workgroup
#define MAX_WGS (T_TOK / TOK_WG + N_EXP)     // 264 (padded expert segments)

// Workspace layout (bytes)
#define WS_META 0               // int meta[128]: counts[0..7], aoff[8..16], cursor[17..24], nwg[25]
#define WS_TOKIDX 512           // int tokidx[T + 32*E] = 33792 B
#define WS_WGU 36864            // bf16 swizzled gate_up: E*8*128*256 uints = 8388608 B
#define WS_WD (36864 + 8388608) // bf16 swizzled down:   E*32*16*256 uints = 4194304 B

typedef __attribute__((ext_vector_type(16))) __bf16 v16bf;
typedef __attribute__((ext_vector_type(8))) float v8f;

static __device__ __forceinline__ unsigned int f2bf(float f) {
  unsigned int u = __float_as_uint(f);
  return (u + 0x7fffu + ((u >> 16) & 1u)) >> 16; // RNE bf16
}

static __device__ __forceinline__ v16bf load_frag_g(const unsigned int* p) {
  union { unsigned int u[8]; v16bf v; } f;
  const uint4* q = (const uint4*)p;
  uint4 a = q[0], b = q[1];
  f.u[0] = a.x; f.u[1] = a.y; f.u[2] = a.z; f.u[3] = a.w;
  f.u[4] = b.x; f.u[5] = b.y; f.u[6] = b.z; f.u[7] = b.w;
  return f.v;
}

static __device__ __forceinline__ v16bf frag_from_regs(const unsigned int* u) {
  union { unsigned int u[8]; v16bf v; } f;
#pragma unroll
  for (int i = 0; i < 8; i++) f.u[i] = u[i];
  return f.v;
}

// A-matrix 16x32 bf16 layout: packed-pair K offset for VGPR j, lane-half h
static __device__ __forceinline__ int a_koff(int j, int h) {
  return (j < 4) ? (h * 8 + 2 * j) : (16 + h * 8 + 2 * (j - 4));
}

// ---------------------------------------------------------------------------
// Pass 0: init meta + token index padding
// ---------------------------------------------------------------------------
__global__ void k_init(int* meta, int* tokidx) {
  int i = blockIdx.x * blockDim.x + threadIdx.x;
  if (i < T_TOK + TOK_WG * N_EXP) tokidx[i] = -1;
  if (i < 32) meta[i] = 0;
}

// Pass 1: histogram of tokens per expert
__global__ void k_histo(const int* __restrict__ ids, int* meta) {
  int t = blockIdx.x * blockDim.x + threadIdx.x;
  if (t < T_TOK) atomicAdd(&meta[ids[t] & (N_EXP - 1)], 1);
}

// Pass 2: 32-aligned exclusive scan (tiny, single thread)
__global__ void k_scan(int* meta) {
  int acc = 0;
#pragma unroll
  for (int e = 0; e < N_EXP; e++) {
    meta[8 + e] = acc;       // aligned segment start
    meta[17 + e] = acc;      // scatter cursor
    acc += (meta[e] + TOK_WG - 1) & ~(TOK_WG - 1);
  }
  meta[16] = acc;            // aligned end
  meta[25] = acc / TOK_WG;   // number of live workgroups
}

// Pass 3: scatter token ids into expert-sorted order
__global__ void k_scatter(const int* __restrict__ ids, int* meta, int* tokidx) {
  int t = blockIdx.x * blockDim.x + threadIdx.x;
  if (t < T_TOK) {
    int e = ids[t] & (N_EXP - 1);
    int pos = atomicAdd(&meta[17 + e], 1);
    tokidx[pos] = t;
  }
}

// ---------------------------------------------------------------------------
// Weight pre-swizzle: fp32 -> bf16 in per-lane WMMA B-fragment order.
// B-matrix 32x16 bf16 layout: lane n<16 -> N=n, VGPR j<4 holds K=2j,2j+1, j>=4 K=2j+8,2j+9;
// lane n>=16 -> N=n-16, same K plus 8.  Tile = 256 uints, lane-contiguous.
// ---------------------------------------------------------------------------
__global__ void k_swz_wgu(const float* __restrict__ gu, unsigned int* __restrict__ wguB) {
  int idx = blockIdx.x * blockDim.x + threadIdx.x; // E*8*128*256 = 2097152
  int u = idx & 255, j = u & 7, lane = u >> 3;
  int tile = idx >> 8;
  int ct = tile & 127, kk = (tile >> 7) & 7, e = tile >> 10;
  int h = lane >> 4, n = lane & 15;
  int k = kk * 32 + ((j < 4) ? (h * 8 + 2 * j) : (16 + h * 8 + 2 * (j - 4)));
  int col = ct * 16 + n;
  const float* src = gu + (size_t)e * L_DIM * (2 * F_DIM) + (size_t)k * (2 * F_DIM) + col;
  wguB[idx] = f2bf(src[0]) | (f2bf(src[2 * F_DIM]) << 16);
}

__global__ void k_swz_wd(const float* __restrict__ dw, unsigned int* __restrict__ wdB) {
  int idx = blockIdx.x * blockDim.x + threadIdx.x; // E*32*16*256 = 1048576
  int u = idx & 255, j = u & 7, lane = u >> 3;
  int tile = idx >> 8;
  int nt = tile & 15, kk = (tile >> 4) & 31, e = tile >> 9;
  int h = lane >> 4, n = lane & 15;
  int k = kk * 32 + ((j < 4) ? (h * 8 + 2 * j) : (16 + h * 8 + 2 * (j - 4)));
  int col = nt * 16 + n;
  const float* src = dw + (size_t)e * F_DIM * L_DIM + (size_t)k * L_DIM + col;
  wdB[idx] = f2bf(src[0]) | (f2bf(src[L_DIM]) << 16);
}

// ---------------------------------------------------------------------------
// Fused MoE kernel: 32 tokens (2 WMMA M-tiles, one expert) per workgroup,
// 8 waves. Each B fragment is reused by both M-tiles -> 2x L2-traffic cut.
// GEMM1 (32x2048x256) -> SwiGLU -> LDS -> GEMM2 (32x256x1024) -> out
// ---------------------------------------------------------------------------
#define SX_STRIDE 132   // uints per x row   (264 bf16, bank-conflict padded)
#define SA_STRIDE 516   // uints per act row (1032 bf16, bank-conflict padded)

__launch_bounds__(256, 1)
__global__ void k_moe(const float* __restrict__ x, float* __restrict__ out,
                      const unsigned int* __restrict__ wguB,
                      const unsigned int* __restrict__ wdB,
                      const int* __restrict__ meta, const int* __restrict__ tokidx) {
  __shared__ int sTok[TOK_WG];
  __shared__ unsigned int sX[TOK_WG * SX_STRIDE]; // x tile, bf16 pairs (16.9 KB)
  extern __shared__ unsigned int sAct[];          // act tile, bf16 pairs (66 KB dynamic)

  int wg = blockIdx.x;
  if (wg >= meta[25]) return;
  int rowBase = wg * TOK_WG;

  int e = 0;
#pragma unroll
  for (int i = 1; i < N_EXP; i++)
    if (rowBase >= meta[8 + i]) e = i;

  int tid = threadIdx.x;
  if (tid < TOK_WG) sTok[tid] = tokidx[rowBase + tid];
  __syncthreads();

  // Stage gathered x rows into LDS as packed bf16 (row-major, padded stride).
  for (int i = tid; i < TOK_WG * (L_DIM / 2); i += 256) {
    int r = i >> 7, c = i & 127;
    int tok = sTok[r];
    unsigned int v = 0;
    if (tok >= 0) {
      const float* p = x + (size_t)tok * L_DIM + c * 2;
      v = f2bf(p[0]) | (f2bf(p[1]) << 16);
    }
    sX[r * SX_STRIDE + c] = v;
  }
  __syncthreads();

  int lane = tid & 31, wave = tid >> 5;
  int h = lane >> 4, m = lane & 15;

  // Preload A fragments for both M-tiles (2 x 8 K-steps) into VGPRs.
  unsigned int aX[M_TILES][8][8];
#pragma unroll
  for (int mt = 0; mt < M_TILES; mt++)
#pragma unroll
    for (int kk = 0; kk < 8; kk++)
#pragma unroll
      for (int j = 0; j < 8; j++)
        aX[mt][kk][j] =
            sX[(mt * TILE_M + m) * SX_STRIDE + ((kk * 32 + a_koff(j, h)) >> 1)];

  // ---------------- GEMM1: h = X @ Wgu, fused SwiGLU ----------------
  const unsigned int* wguE = wguB + (size_t)e * (8 * 128 * 256);
  v8f vzero = {};
#pragma unroll 1
  for (int t8 = 0; t8 < 8; t8++) {
    int ct = wave + t8 * 8; // gate col-tile in [0,64); up tile = ct+64
    v8f accg[M_TILES], accu[M_TILES];
#pragma unroll
    for (int mt = 0; mt < M_TILES; mt++) { accg[mt] = vzero; accu[mt] = vzero; }
    if (t8 < 7)
      __builtin_prefetch(wguE + ((size_t)(ct + 8)) * 256 + lane * 8, 0, 1);
#pragma unroll
    for (int kk = 0; kk < 8; kk++) {
      v16bf bg = load_frag_g(wguE + ((size_t)kk * 128 + ct) * 256 + lane * 8);
      v16bf bu = load_frag_g(wguE + ((size_t)kk * 128 + (ct + 64)) * 256 + lane * 8);
#pragma unroll
      for (int mt = 0; mt < M_TILES; mt++) {
        v16bf a = frag_from_regs(aX[mt][kk]);
        accg[mt] = __builtin_amdgcn_wmma_f32_16x16x32_bf16(false, a, false, bg, (short)0, accg[mt], false, false);
        accu[mt] = __builtin_amdgcn_wmma_f32_16x16x32_bf16(false, a, false, bu, (short)0, accu[mt], false, false);
      }
    }
    // SwiGLU: silu(gate)*up, write act tile to LDS as bf16.
#pragma unroll
    for (int mt = 0; mt < M_TILES; mt++)
#pragma unroll
      for (int r = 0; r < 8; r++) {
        float g = accg[mt][r], uu = accu[mt][r];
        float s = g / (1.0f + __expf(-g));
        int row = mt * TILE_M + r + h * 8;
        int col = ct * 16 + m;
        ((unsigned short*)sAct)[row * (2 * SA_STRIDE) + col] = (unsigned short)f2bf(s * uu);
      }
  }
  __syncthreads();

  // ---------------- GEMM2: y = act @ Wd ----------------
  const unsigned int* wdE = wdB + (size_t)e * (32 * 16 * 256);
  int nt0 = wave, nt1 = wave + 8;
  v8f acc[M_TILES][2];
#pragma unroll
  for (int mt = 0; mt < M_TILES; mt++) { acc[mt][0] = vzero; acc[mt][1] = vzero; }
#pragma unroll 2
  for (int kk = 0; kk < 32; kk++) {
    unsigned int au[M_TILES][8];
#pragma unroll
    for (int mt = 0; mt < M_TILES; mt++)
#pragma unroll
      for (int j = 0; j < 8; j++)
        au[mt][j] =
            sAct[(mt * TILE_M + m) * SA_STRIDE + ((kk * 32 + a_koff(j, h)) >> 1)];
    v16bf b0 = load_frag_g(wdE + ((size_t)kk * 16 + nt0) * 256 + lane * 8);
    v16bf b1 = load_frag_g(wdE + ((size_t)kk * 16 + nt1) * 256 + lane * 8);
#pragma unroll
    for (int mt = 0; mt < M_TILES; mt++) {
      v16bf a = frag_from_regs(au[mt]);
      acc[mt][0] = __builtin_amdgcn_wmma_f32_16x16x32_bf16(false, a, false, b0, (short)0, acc[mt][0], false, false);
      acc[mt][1] = __builtin_amdgcn_wmma_f32_16x16x32_bf16(false, a, false, b1, (short)0, acc[mt][1], false, false);
    }
  }

  // Scatter fp32 results directly (each token owned by exactly one row).
#pragma unroll
  for (int mt = 0; mt < M_TILES; mt++)
#pragma unroll
    for (int r = 0; r < 8; r++) {
      int row = mt * TILE_M + r + h * 8;
      int tok = sTok[row];
      if (tok >= 0) {
        out[(size_t)tok * L_DIM + nt0 * 16 + m] = acc[mt][0][r];
        out[(size_t)tok * L_DIM + nt1 * 16 + m] = acc[mt][1][r];
      }
    }
}

// ---------------------------------------------------------------------------
extern "C" void kernel_launch(void* const* d_in, const int* in_sizes, int n_in,
                              void* d_out, int out_size, void* d_ws, size_t ws_size,
                              hipStream_t stream) {
  const float* x = (const float*)d_in[0];
  const float* gu = (const float*)d_in[1];
  const float* dw = (const float*)d_in[2];
  const int* ids = (const int*)d_in[3];
  float* out = (float*)d_out;

  char* ws = (char*)d_ws;
  int* meta = (int*)(ws + WS_META);
  int* tokidx = (int*)(ws + WS_TOKIDX);
  unsigned int* wguB = (unsigned int*)(ws + WS_WGU);
  unsigned int* wdB = (unsigned int*)(ws + WS_WD);

  (void)in_sizes; (void)n_in; (void)out_size; (void)ws_size;

  k_init<<<(T_TOK + TOK_WG * N_EXP + 255) / 256, 256, 0, stream>>>(meta, tokidx);
  k_swz_wgu<<<(N_EXP * 8 * 128 * 256) / 256, 256, 0, stream>>>(gu, wguB);
  k_swz_wd<<<(N_EXP * 32 * 16 * 256) / 256, 256, 0, stream>>>(dw, wdB);
  k_histo<<<T_TOK / 256, 256, 0, stream>>>(ids, meta);
  k_scan<<<1, 1, 0, stream>>>(meta);
  k_scatter<<<T_TOK / 256, 256, 0, stream>>>(ids, meta, tokidx);
  k_moe<<<MAX_WGS, 256, (size_t)TOK_WG * SA_STRIDE * 4, stream>>>(x, out, wguB, wdB, meta, tokidx);
}
